// Transformer_63453846831524
// MI455X (gfx1250) — matrix operand order repaired and checked by reading, not compiled, not run on hardware
//
#include <hip/hip_runtime.h>
#include <hip/hip_bf16.h>

typedef __attribute__((ext_vector_type(16))) _Float16 v16h;
typedef __attribute__((ext_vector_type(8)))  float    v8f;

#define TOKENS 2048   // B*T = 16*128
#define CDIM   256
#define NHEAD  4
#define HDIM   64
#define TSEQ   128
#define VOCAB  50257
#define VOCABP 50304  // padded to multiple of 128

// padded LDS tile strides (halves); multiples of 8 halves keep 16B alignment
#define AS_STRIDE 40   // 64 rows x 32 halves, 80B row pitch
#define BS_STRIDE 136  // 32 rows x 128 halves, 272B row pitch

// ---------------------------------------------------------------------------
// gfx1250 async global->LDS copy (ASYNCcnt-tracked), guarded so compile never
// breaks if the builtin is absent on this toolchain.
#if defined(__gfx1250__) && __has_builtin(__builtin_amdgcn_global_load_async_to_lds_b128) && __has_builtin(__builtin_amdgcn_s_wait_asynccnt)
#define HAS_ASYNC_LDS 1
#else
#define HAS_ASYNC_LDS 0
#endif

#if HAS_ASYNC_LDS
// builtin signature (from hipcc diagnostic): param0 is int4 in AS1 (global),
// param1 int4 in AS3 (LDS), then imm offset, imm cpol.
typedef int v4i_gcc __attribute__((vector_size(16)));
typedef __attribute__((address_space(1))) v4i_gcc* as1_v4i;
typedef __attribute__((address_space(3))) v4i_gcc* as3_v4i;
#endif

// copy 16 bytes global -> LDS (both 16B aligned)
__device__ __forceinline__ void cp16(_Float16* lds_dst, const _Float16* g_src) {
#if HAS_ASYNC_LDS
    __builtin_amdgcn_global_load_async_to_lds_b128(
        (as1_v4i)g_src, (as3_v4i)lds_dst, 0, 0);
#else
    *(float4*)lds_dst = *(const float4*)g_src;
#endif
}
// wait until at most `n` of this wave's async copies remain in flight
template <int NREM>
__device__ __forceinline__ void cp_wait() {
#if HAS_ASYNC_LDS
    __builtin_amdgcn_s_wait_asynccnt(NREM);
#endif
}

// A-matrix (16x32 f16) element i -> k index (ISA 7.12.2: 8-k groups
// interleaved between half-waves)
__device__ __forceinline__ int kmapA(int i, int half) {
    return (i & 7) + ((i >> 3) << 4) + (half << 3);
}
// B-matrix (32x16 f16): same k order as A (8-k groups interleaved between half-waves)
__device__ __forceinline__ int kmapB(int i, int half) {
    return (i & 7) + ((i >> 3) << 4) + (half << 3);
}

__device__ __forceinline__ v8f wmma_f16(v16h a, v16h b, v8f c) {
    return __builtin_amdgcn_wmma_f32_16x16x32_f16(false, a, false, b,
                                                  (short)0, c, false, false);
}

// ---------------------------------------------------------------------------
// Weight pre-conversion kernels (run once per launch; deterministic)

// Wq/Wk/Wv [L,H,C,HD] f32 -> [L][C][H*HD] f16 (fused head layout)
__global__ __launch_bounds__(256) void cvt_qkv_kernel(
    const float* __restrict__ Wq, const float* __restrict__ Wk,
    const float* __restrict__ Wv, _Float16* __restrict__ Q16,
    _Float16* __restrict__ K16, _Float16* __restrict__ V16) {
    int s = blockIdx.x * 256 + threadIdx.x;     // over L*H*C*HD = 262144
    int d = s & 63;
    int c = (s >> 6) & 255;
    int h = (s >> 14) & 3;
    int l = s >> 16;
    int dst = (l * 256 + c) * 256 + h * 64 + d;
    Q16[dst] = (_Float16)Wq[s];
    K16[dst] = (_Float16)Wk[s];
    V16[dst] = (_Float16)Wv[s];
}

__global__ __launch_bounds__(256) void cvt_f16_kernel(
    const float* __restrict__ src, _Float16* __restrict__ dst, int n) {
    int s = blockIdx.x * 256 + threadIdx.x;
    if (s < n) dst[s] = (_Float16)src[s];
}

// Whead [C,V] f32 -> [C,VOCABP] f16, zero padded tail
__global__ __launch_bounds__(256) void cvt_head_kernel(
    const float* __restrict__ W, _Float16* __restrict__ D) {
    int r = blockIdx.y;
    int c = blockIdx.x * 256 + threadIdx.x;
    if (c < VOCABP)
        D[r * VOCABP + c] = (c < VOCAB) ? (_Float16)W[r * VOCAB + c] : (_Float16)0.0f;
}

// ---------------------------------------------------------------------------
__global__ __launch_bounds__(256) void embed_kernel(
    const int* __restrict__ idx, const float* __restrict__ tok,
    const float* __restrict__ pos, float* __restrict__ x) {
    int token = blockIdx.x, c = threadIdx.x;
    int id = idx[token];
    int t  = token & (TSEQ - 1);
    x[token * CDIM + c] = tok[id * CDIM + c] + pos[t * CDIM + c];
}

// LayerNorm over C=256 (f32 in, f16 out), one block per token
__global__ __launch_bounds__(256) void ln_kernel(
    const float* __restrict__ x, const float* __restrict__ g,
    const float* __restrict__ b, _Float16* __restrict__ y) {
    __shared__ float red[256];
    int t = blockIdx.x, c = threadIdx.x;
    float v = x[t * CDIM + c];
    red[c] = v;
    __syncthreads();
    #pragma unroll
    for (int s = 128; s > 0; s >>= 1) {
        if (c < s) red[c] += red[c + s];
        __syncthreads();
    }
    float mu = red[0] * (1.0f / CDIM);
    __syncthreads();
    float d = v - mu;
    red[c] = d * d;
    __syncthreads();
    #pragma unroll
    for (int s = 128; s > 0; s >>= 1) {
        if (c < s) red[c] += red[c + s];
        __syncthreads();
    }
    float var = red[0] * (1.0f / CDIM);
    y[t * CDIM + c] = (_Float16)(d * rsqrtf(var + 1e-5f) * g[c] + b[c]);
}

// ---------------------------------------------------------------------------
// C[M,N] = A[M,K] @ B[K,N] (+bias)(ReLU)(+resid). A,B f16 in global; output
// f16 (OUTH) or f32. Block tile 64x128, K-step 32; 8 waves in 4(m)x2(n),
// 4 accumulators/wave. Double-buffered LDS tiles filled by ASYNCcnt-tracked
// async copies: tile i+1 copies overlap tile i WMMA; each thread issues 3
// async b128 ops per tile, so s_wait_asynccnt 3 drains exactly tile i.
template <bool RELU, bool RESID, bool OUTH>
__global__ __launch_bounds__(256) void gemm_wmma(
    const _Float16* __restrict__ A, int lda,
    const _Float16* __restrict__ B, int ldb,
    const float* __restrict__ bias,
    const float* __restrict__ R,
    void* __restrict__ Cout, int ldc,
    int M, int N, int K) {
    __shared__ __align__(16) _Float16 As[2][64 * AS_STRIDE];  // 2 x 5 KB
    __shared__ __align__(16) _Float16 Bs[2][32 * BS_STRIDE];  // 2 x 8.5 KB
    const int tid = threadIdx.x;
    const int w = tid >> 5, lane = tid & 31;
    const int half = lane >> 4, l15 = lane & 15;
    const int wm = w >> 1, wn = w & 1;
    const int gm0 = blockIdx.y * 64;
    const int gn0 = blockIdx.x * 128;

    // per-thread staging coordinates (16B chunks)
    const int ra = tid >> 2, ca = (tid & 3) * 8;       // A: 64x32 = 256 chunks
    const int rb0 = tid >> 4, cb0 = (tid & 15) * 8;    // B: 32x128 = 512 chunks
    const int rb1 = (tid + 256) >> 4, cb1 = cb0;

    v8f acc[4] = {};

    const int nk = K >> 5;
    // prologue: stage tile 0 into buffer 0
    cp16(&As[0][ra * AS_STRIDE + ca], &A[(gm0 + ra) * lda + ca]);
    cp16(&Bs[0][rb0 * BS_STRIDE + cb0], &B[rb0 * ldb + gn0 + cb0]);
    cp16(&Bs[0][rb1 * BS_STRIDE + cb1], &B[rb1 * ldb + gn0 + cb1]);

    for (int it = 0; it < nk; ++it) {
        const int buf = it & 1;
        if (it + 1 < nk) {
            // stage next tile into the other buffer (overlaps this tile's WMMA)
            const int k0 = (it + 1) << 5;
            cp16(&As[buf ^ 1][ra * AS_STRIDE + ca], &A[(gm0 + ra) * lda + k0 + ca]);
            cp16(&Bs[buf ^ 1][rb0 * BS_STRIDE + cb0], &B[(k0 + rb0) * ldb + gn0 + cb0]);
            cp16(&Bs[buf ^ 1][rb1 * BS_STRIDE + cb1], &B[(k0 + rb1) * ldb + gn0 + cb1]);
            if (it + 2 < nk)  // prefetch tile i+2 B rows toward L2
                __builtin_prefetch(&B[((it + 2) << 5) * ldb + gn0], 0, 1);
            cp_wait<3>();   // drain tile it copies; leave tile it+1 in flight
        } else {
            cp_wait<0>();
        }
        __syncthreads();    // tile `buf` fully visible to all waves

        v16h a;
        #pragma unroll
        for (int i = 0; i < 16; ++i)
            a[i] = As[buf][(16 * wm + l15) * AS_STRIDE + kmapA(i, half)];
        #pragma unroll
        for (int j = 0; j < 4; ++j) {
            v16h bf;
            #pragma unroll
            for (int i = 0; i < 16; ++i)
                bf[i] = Bs[buf][kmapB(i, half) * BS_STRIDE + 64 * wn + 16 * j + l15];
            acc[j] = wmma_f16(a, bf, acc[j]);
        }
        __syncthreads();    // all reads of `buf` done before it is restaged
    }

    // epilogue: vgpr r holds row r+8*half, col = lane&15 (+16j +64wn)
    #pragma unroll
    for (int j = 0; j < 4; ++j) {
        #pragma unroll
        for (int r = 0; r < 8; ++r) {
            int row = gm0 + 16 * wm + r + 8 * half;
            int col = gn0 + 64 * wn + 16 * j + l15;
            if (col < N) {
                float v = acc[j][r];
                if (bias) v += bias[col];
                if (RELU) v = fmaxf(v, 0.0f);
                if (RESID) v += R[row * ldc + col];
                if (OUTH) ((_Float16*)Cout)[row * ldc + col] = (_Float16)v;
                else      ((float*)Cout)[row * ldc + col] = v;
            }
        }
    }
}

// ---------------------------------------------------------------------------
// Fused causal attention for one (batch, head). q/k/v are f16 [2048,256]
// head-concat buffers; output xatt f16 same layout. One block per (b,h).
__global__ __launch_bounds__(256) void attn_kernel(
    const _Float16* __restrict__ q, const _Float16* __restrict__ k,
    const _Float16* __restrict__ v, _Float16* __restrict__ o) {
    __shared__ __align__(16) _Float16 smem[3 * TSEQ * HDIM];  // 48 KB
    _Float16* Qh = smem;
    _Float16* Kh = smem + TSEQ * HDIM;
    _Float16* Vh = smem + 2 * TSEQ * HDIM;
    _Float16* Ph = smem;  // 128x128 halves, reuses Q+K region after barrier

    const int bh = blockIdx.x;          // 0..63
    const int bb = bh >> 2, hh = bh & 3;
    const int tok0 = bb * TSEQ, hoff = hh * HDIM;
    const int tid = threadIdx.x;
    const int w = tid >> 5, lane = tid & 31;
    const int half = lane >> 4, l15 = lane & 15;
    const int m0 = 16 * w;

    // stage Q,K,V: 3 x (128 rows x 64 halves) = 3 x 1024 chunks of 16B
    #pragma unroll
    for (int i = 0; i < 4; ++i) {
        int c = tid + i * 256;                 // 0..1023
        int r = c >> 3, c8 = (c & 7) * 8;
        int g = (tok0 + r) * CDIM + hoff + c8;
        cp16(&Qh[r * HDIM + c8], &q[g]);
        cp16(&Kh[r * HDIM + c8], &k[g]);
        cp16(&Vh[r * HDIM + c8], &v[g]);
    }
    cp_wait<0>();
    __syncthreads();

    // phase 1: S[16 x 128] for this wave's rows, in registers
    v8f acc[8] = {};
    #pragma unroll
    for (int d0 = 0; d0 < HDIM; d0 += 32) {
        v16h a;
        #pragma unroll
        for (int i = 0; i < 16; ++i)
            a[i] = Qh[(m0 + l15) * HDIM + d0 + kmapA(i, half)];
        #pragma unroll
        for (int j = 0; j < 8; ++j) {
            v16h bf;
            #pragma unroll
            for (int i = 0; i < 16; ++i)  // B[k=d][n=s] = K[s][d]
                bf[i] = Kh[(16 * j + l15) * HDIM + d0 + kmapB(i, half)];
            acc[j] = wmma_f16(a, bf, acc[j]);
        }
    }
    __syncthreads();  // all waves done with Q/K before P overwrites them

    // causal mask + softmax; write P (f16) into LDS
    const float scale = 0.0625f;  // C^-0.5 = 1/16 (reference scales by n_embd)
    #pragma unroll
    for (int r = 0; r < 8; ++r) {
        int rl = r + 8 * half;
        int grow = m0 + rl;
        float vals[8];
        #pragma unroll
        for (int j = 0; j < 8; ++j) {
            int col = 16 * j + l15;
            float s = acc[j][r] * scale;
            vals[j] = (col <= grow) ? s : -__builtin_inff();
        }
        float mx = vals[0];
        #pragma unroll
        for (int j = 1; j < 8; ++j) mx = fmaxf(mx, vals[j]);
        #pragma unroll
        for (int d = 1; d < 16; d <<= 1) mx = fmaxf(mx, __shfl_xor(mx, d));
        float sum = 0.0f;
        #pragma unroll
        for (int j = 0; j < 8; ++j) { vals[j] = __expf(vals[j] - mx); sum += vals[j]; }
        #pragma unroll
        for (int d = 1; d < 16; d <<= 1) sum += __shfl_xor(sum, d);
        float inv = 1.0f / sum;
        #pragma unroll
        for (int j = 0; j < 8; ++j)
            Ph[(m0 + rl) * TSEQ + 16 * j + l15] = (_Float16)(vals[j] * inv);
    }

    // phase 2: O[16 x 64] = P[16 x 128] @ V[128 x 64]
    v8f oo[4] = {};
    #pragma unroll
    for (int s0 = 0; s0 < TSEQ; s0 += 32) {
        v16h a;
        #pragma unroll
        for (int i = 0; i < 16; ++i)
            a[i] = Ph[(m0 + l15) * TSEQ + s0 + kmapA(i, half)];
        #pragma unroll
        for (int j = 0; j < 4; ++j) {
            v16h bf;
            #pragma unroll
            for (int i = 0; i < 16; ++i)
                bf[i] = Vh[(s0 + kmapB(i, half)) * HDIM + 16 * j + l15];
            oo[j] = wmma_f16(a, bf, oo[j]);
        }
    }

    #pragma unroll
    for (int j = 0; j < 4; ++j) {
        #pragma unroll
        for (int r = 0; r < 8; ++r) {
            int rl = r + 8 * half;
            o[(tok0 + m0 + rl) * CDIM + hoff + 16 * j + l15] = (_Float16)oo[j][r];
        }
    }
}

// ---------------------------------------------------------------------------
extern "C" void kernel_launch(void* const* d_in, const int* in_sizes, int n_in,
                              void* d_out, int out_size, void* d_ws, size_t ws_size,
                              hipStream_t stream) {
    const int*   idx     = (const int*)  d_in[0];
    const float* tok_emb = (const float*)d_in[1];
    const float* pos_emb = (const float*)d_in[2];
    const float* Wq      = (const float*)d_in[3];   // [L,H,C,HD]
    const float* Wk      = (const float*)d_in[4];
    const float* Wv      = (const float*)d_in[5];
    const float* Wp      = (const float*)d_in[6];   // [L,C,C]
    const float* bp      = (const float*)d_in[7];
    const float* ln1_g   = (const float*)d_in[8];
    const float* ln1_b   = (const float*)d_in[9];
    const float* ln2_g   = (const float*)d_in[10];
    const float* ln2_b   = (const float*)d_in[11];
    const float* W1      = (const float*)d_in[12];  // [L,C,4C]
    const float* b1      = (const float*)d_in[13];
    const float* W2      = (const float*)d_in[14];  // [L,4C,C]
    const float* b2      = (const float*)d_in[15];
    const float* lnf_g   = (const float*)d_in[16];
    const float* lnf_b   = (const float*)d_in[17];
    const float* Whead   = (const float*)d_in[18];  // [C,V]
    const float* bhead   = (const float*)d_in[19];
    float* out = (float*)d_out;

    const size_t NC = (size_t)TOKENS * CDIM;  // 524288
    char* p = (char*)d_ws;
    auto carve = [&](size_t bytes) {
        void* r = (void*)p;
        p += (bytes + 255) & ~(size_t)255;
        return r;
    };
    float*     x     = (float*)    carve(NC * 4);
    _Float16*  hbuf  = (_Float16*) carve(NC * 2);
    _Float16*  qb    = (_Float16*) carve(NC * 2);
    _Float16*  kb    = (_Float16*) carve(NC * 2);
    _Float16*  vb    = (_Float16*) carve(NC * 2);
    _Float16*  xatt  = (_Float16*) carve(NC * 2);
    _Float16*  mlp   = (_Float16*) carve((size_t)TOKENS * 1024 * 2);
    _Float16*  Wq16  = (_Float16*) carve((size_t)4 * 65536 * 2);
    _Float16*  Wk16  = (_Float16*) carve((size_t)4 * 65536 * 2);
    _Float16*  Wv16  = (_Float16*) carve((size_t)4 * 65536 * 2);
    _Float16*  Wp16  = (_Float16*) carve((size_t)4 * 65536 * 2);
    _Float16*  W116  = (_Float16*) carve((size_t)4 * 262144 * 2);
    _Float16*  W216  = (_Float16*) carve((size_t)4 * 262144 * 2);
    _Float16*  Wh16  = (_Float16*) carve((size_t)CDIM * VOCABP * 2);

    // ---- one-time (per launch) weight conversion to f16 ----
    cvt_qkv_kernel<<<1024, 256, 0, stream>>>(Wq, Wk, Wv, Wq16, Wk16, Wv16);
    cvt_f16_kernel<<<1024, 256, 0, stream>>>(Wp, Wp16, 4 * 65536);
    cvt_f16_kernel<<<4096, 256, 0, stream>>>(W1, W116, 4 * 262144);
    cvt_f16_kernel<<<4096, 256, 0, stream>>>(W2, W216, 4 * 262144);
    cvt_head_kernel<<<dim3(VOCABP / 256, CDIM), 256, 0, stream>>>(Whead, Wh16);

    embed_kernel<<<TOKENS, 256, 0, stream>>>(idx, tok_emb, pos_emb, x);

    for (int l = 0; l < 4; ++l) {
        ln_kernel<<<TOKENS, 256, 0, stream>>>(x, ln1_g + 256 * l, ln1_b + 256 * l, hbuf);

        // fused-head QKV: B16 = [C, H*HD], output already concat-head layout
        gemm_wmma<false, false, true><<<dim3(2, TOKENS / 64), 256, 0, stream>>>(
            hbuf, CDIM, Wq16 + l * 65536, CDIM, nullptr, nullptr,
            qb, CDIM, TOKENS, CDIM, CDIM);
        gemm_wmma<false, false, true><<<dim3(2, TOKENS / 64), 256, 0, stream>>>(
            hbuf, CDIM, Wk16 + l * 65536, CDIM, nullptr, nullptr,
            kb, CDIM, TOKENS, CDIM, CDIM);
        gemm_wmma<false, false, true><<<dim3(2, TOKENS / 64), 256, 0, stream>>>(
            hbuf, CDIM, Wv16 + l * 65536, CDIM, nullptr, nullptr,
            vb, CDIM, TOKENS, CDIM, CDIM);

        attn_kernel<<<16 * NHEAD, 256, 0, stream>>>(qb, kb, vb, xatt);

        // x = x + attn_out @ Wp + bp
        gemm_wmma<false, true, false><<<dim3(2, TOKENS / 64), 256, 0, stream>>>(
            xatt, CDIM, Wp16 + l * 65536, CDIM, bp + 256 * l, x,
            x, CDIM, TOKENS, CDIM, CDIM);

        ln_kernel<<<TOKENS, 256, 0, stream>>>(x, ln2_g + 256 * l, ln2_b + 256 * l, hbuf);

        // mlp = relu(h2 @ W1 + b1)
        gemm_wmma<true, false, true><<<dim3(8, TOKENS / 64), 256, 0, stream>>>(
            hbuf, CDIM, W116 + l * 262144, 1024, b1 + 1024 * l, nullptr,
            mlp, 1024, TOKENS, 1024, CDIM);

        // x = x + mlp @ W2 + b2
        gemm_wmma<false, true, false><<<dim3(2, TOKENS / 64), 256, 0, stream>>>(
            mlp, 1024, W216 + l * 262144, CDIM, b2 + 256 * l, x,
            x, CDIM, TOKENS, CDIM, 1024);
    }

    ln_kernel<<<TOKENS, 256, 0, stream>>>(x, lnf_g, lnf_b, hbuf);

    // logits = xf @ Wh16(padded ldb) + bhead -> d_out [2048, 50257]
    gemm_wmma<false, false, false><<<dim3(VOCABP / 128, TOKENS / 64), 256, 0, stream>>>(
        hbuf, CDIM, Wh16, VOCABP, bhead, nullptr,
        out, VOCAB, TOKENS, VOCAB, CDIM);
}
